// YOLOV5TorchObjectDetector_58540404244891
// MI455X (gfx1250) — compile-verified, matrix operand order checked by reading, not run on hardware
//
#include <hip/hip_runtime.h>
#include <hip/hip_bf16.h>
#include <stdint.h>

namespace {
constexpr int   B        = 16;
constexpr int   N        = 25200;
constexpr int   NC       = 80;
constexpr int   C        = NC + 5;     // 85 floats per row
constexpr int   K        = 1024;       // MAX_NMS
constexpr int   MAX_DET  = 300;
constexpr int   OUT_D    = 6 + NC;     // 86
constexpr float CONF     = 0.4f;
constexpr float IOU_TH   = 0.45f;
constexpr float MAX_WH   = 4096.0f;
constexpr float SCL      = 1.0f / 512.0f;  // uniform scale; IoU is scale-invariant,
                                           // keeps offset boxes/areas inside f16 range
constexpr int   ROWS_PER_TILE = 128;
constexpr int   TILES_PER_IMG = (N + ROWS_PER_TILE - 1) / ROWS_PER_TILE; // 197
constexpr int   SORT_CAP      = 4096;
}

typedef __attribute__((ext_vector_type(16))) _Float16 v16h;
typedef __attribute__((ext_vector_type(8)))  float    v8f;
typedef int v4i __attribute__((vector_size(16)));   // matches async-LDS builtin param

#if defined(__AMDGCN__) && __has_builtin(__builtin_amdgcn_global_load_async_to_lds_b128)
#define USE_ASYNC_LDS 1
#else
#define USE_ASYNC_LDS 0
#endif

// ---------------------------------------------------------------- zero / init
__global__ void k_zero(float* out, unsigned* mask, int* posOf) {
  size_t i = (size_t)blockIdx.x * blockDim.x + threadIdx.x;
  size_t stride = (size_t)gridDim.x * blockDim.x;
  const size_t no = (size_t)B * MAX_DET * OUT_D;
  for (size_t p = i; p < no; p += stride) out[p] = 0.0f;
  const size_t nm = (size_t)B * K * 32;
  for (size_t p = i; p < nm; p += stride) mask[p] = 0u;
  const size_t np = (size_t)B * K;
  for (size_t p = i; p < np; p += stride) posOf[p] = -1;
}

// ------------------------------------------------- stage 1: score scan (137 MB)
// Streams prediction rows through LDS via CDNA5 async global->LDS DMA.
__global__ __launch_bounds__(ROWS_PER_TILE)
void k_score(const float* __restrict__ pred, float* __restrict__ score) {
  __shared__ float tile[ROWS_PER_TILE * C];  // 43,520 B
  int bid  = blockIdx.x;
  int img  = bid / TILES_PER_IMG;
  int t    = bid % TILES_PER_IMG;
  int row0 = t * ROWS_PER_TILE;
  int rows = N - row0; if (rows > ROWS_PER_TILE) rows = ROWS_PER_TILE; // 128 or 112 (both %4==0)
  const float* gsrc = pred + ((size_t)img * N + row0) * C;
  int chunks = (rows * C) / 4;  // 16-byte chunks; start address is 16B aligned
  int tid = threadIdx.x;
#if USE_ASYNC_LDS
  for (int c = tid; c < chunks; c += ROWS_PER_TILE) {
    __builtin_amdgcn_global_load_async_to_lds_b128(
        (__attribute__((address_space(1))) v4i*)(gsrc + (size_t)c * 4),
        (__attribute__((address_space(3))) v4i*)(&tile[c * 4]),
        0, 0);
  }
#if __has_builtin(__builtin_amdgcn_s_wait_asynccnt)
  __builtin_amdgcn_s_wait_asynccnt(0);
#else
  asm volatile("s_wait_asynccnt 0" ::: "memory");
#endif
#else
  for (int c = tid; c < chunks; c += ROWS_PER_TILE)
    ((float4*)tile)[c] = ((const float4*)gsrc)[c];
#endif
  __syncthreads();
  if (tid < rows) {
    const float* r = &tile[tid * C];
    float obj = r[4];
    float mc  = r[5];
#pragma unroll 4
    for (int c2 = 1; c2 < NC; ++c2) mc = fmaxf(mc, r[5 + c2]);
    float conf = obj * mc;
    bool valid = (obj > CONF) && (conf > CONF);
    score[(size_t)img * N + row0 + tid] = valid ? conf : -1.0f;
  }
}

// --------------------------------- stage 2: per-image top-1024, sorted descending
__global__ __launch_bounds__(256)
void k_topk(const float* __restrict__ score, int* __restrict__ candIdx) {
  __shared__ unsigned long long s_key[SORT_CAP];  // 32 KB; first 16 KB aliased as hist
  __shared__ int s_cnt, s_t;
  unsigned* hist = (unsigned*)s_key;
  int img = blockIdx.x;
  int tid = threadIdx.x;
  const float* sc = score + (size_t)img * N;
  for (int b = tid; b < 4096; b += 256) hist[b] = 0u;
  __syncthreads();
  for (int n = tid; n < N; n += 256) {
    float s = sc[n];
    if (s >= 0.0f) {
      int bin = (int)(s * 4096.0f);
      bin = bin < 0 ? 0 : (bin > 4095 ? 4095 : bin);
      atomicAdd(&hist[bin], 1u);
    }
  }
  __syncthreads();
  if (tid == 0) {
    int cum = 0, thr = 0;
    for (int b = 4095; b >= 0; --b) {
      cum += (int)hist[b];
      if (cum >= K) { thr = b; break; }
    }
    s_t = thr;
    s_cnt = 0;
  }
  __syncthreads();
  int thr = s_t;
  __syncthreads();  // hist memory reused as keys below
  for (int n = tid; n < N; n += 256) {
    float s = sc[n];
    if (s >= 0.0f) {
      int bin = (int)(s * 4096.0f);
      bin = bin < 0 ? 0 : (bin > 4095 ? 4095 : bin);
      if (bin >= thr) {
        int p = atomicAdd(&s_cnt, 1);
        if (p < SORT_CAP) {
          unsigned sb = __float_as_uint(s);  // s>0 -> monotone bits
          // key: score desc, index asc (matches stable top_k tie order)
          s_key[p] = ((unsigned long long)sb << 32) |
                     (unsigned long long)(0xFFFFFFFFu - (unsigned)n);
        }
      }
    }
  }
  __syncthreads();
  int cnt = s_cnt; if (cnt > SORT_CAP) cnt = SORT_CAP;
  for (int p = tid; p < SORT_CAP; p += 256)
    if (p >= cnt) s_key[p] = 0ULL;
  __syncthreads();
  // bitonic sort, descending
  for (int kk = 2; kk <= SORT_CAP; kk <<= 1) {
    for (int j = kk >> 1; j > 0; j >>= 1) {
      for (int idx = tid; idx < SORT_CAP; idx += 256) {
        int ixj = idx ^ j;
        if (ixj > idx) {
          unsigned long long a = s_key[idx], b = s_key[ixj];
          bool descBlock = ((idx & kk) == 0);
          bool doSwap = descBlock ? (a < b) : (a > b);
          if (doSwap) { s_key[idx] = b; s_key[ixj] = a; }
        }
      }
      __syncthreads();
    }
  }
  for (int p = tid; p < K; p += 256) {
    unsigned long long key = s_key[p];
    int idx = -1;
    if (key != 0ULL) {
      int cand = (int)(0xFFFFFFFFu - (unsigned)(key & 0xFFFFFFFFull));
      if (cand >= 0 && cand < N) idx = cand;
    }
    candIdx[(size_t)img * K + p] = idx;
  }
}

// ----------------------------- stage 3: gather candidate rows, build scaled boxes
__global__ __launch_bounds__(256)
void k_gather(const float* __restrict__ pred, const int* __restrict__ candIdx,
              float* __restrict__ candBox, float* __restrict__ candObb,
              float* __restrict__ candArea, float* __restrict__ candConf,
              float* __restrict__ candCls, int* __restrict__ candValid) {
  int g = blockIdx.x * 256 + threadIdx.x;
  if (g >= B * K) return;
  int img = g / K;
  int idx = candIdx[g];
  float box[4] = {0, 0, 0, 0}, obb[4] = {0, 0, 0, 0};
  float area = 0.f, conf = 0.f;
  int cls = 0, valid = 0;
  if (idx >= 0) {
    const float* r = pred + ((size_t)img * N + idx) * C;
    float obj = r[4];
    float mc = r[5]; int mi = 0;
    for (int c2 = 1; c2 < NC; ++c2) {
      float v = r[5 + c2];
      if (v > mc) { mc = v; mi = c2; }   // argmax: first max index
    }
    conf = obj * mc; cls = mi;
    float cx = r[0], cy = r[1], w = r[2], h = r[3];
    box[0] = cx - 0.5f * w; box[1] = cy - 0.5f * h;
    box[2] = cx + 0.5f * w; box[3] = cy + 0.5f * h;
    float off = (float)cls * MAX_WH;
    obb[0] = (box[0] + off) * SCL; obb[1] = (box[1] + off) * SCL;
    obb[2] = (box[2] + off) * SCL; obb[3] = (box[3] + off) * SCL;
    area = (obb[2] - obb[0]) * (obb[3] - obb[1]);
    valid = ((obj > CONF) && (conf > CONF)) ? 1 : 0;
  }
  candBox[g * 4 + 0] = box[0]; candBox[g * 4 + 1] = box[1];
  candBox[g * 4 + 2] = box[2]; candBox[g * 4 + 3] = box[3];
  candObb[g * 4 + 0] = obb[0]; candObb[g * 4 + 1] = obb[1];
  candObb[g * 4 + 2] = obb[2]; candObb[g * 4 + 3] = obb[3];
  candArea[g] = area; candConf[g] = conf;
  candCls[g] = (float)cls; candValid[g] = valid;
}

// ----------------- stage 4: IoU suppression bitmask; WMMA computes area_i+area_j
// Per wave: one 16x16 tile. S = A*B with A=[area_i, 1, 0...], B=[1, area_j, 0...]^T
// gives S[i][j] = area_i + area_j on the matrix pipe (rank-2 update), co-executing
// with the VALU intersection math. iou>TH  <=>  inter*(1+TH) > TH*S  (no division).
__global__ __launch_bounds__(256)
void k_mask(const float* __restrict__ candObb, const float* __restrict__ candArea,
            unsigned* __restrict__ mask) {
  __shared__ float s_ib[16][4], s_ia[16];
  __shared__ float s_jb[128][4], s_ja[128];
  int bid = blockIdx.x;
  int img = bid >> 9;            // 512 blocks per image (64 row-tiles x 8 col-groups)
  int rem = bid & 511;
  int i0 = (rem >> 3) * 16;
  int j0 = (rem & 7) * 128;
  int tid = threadIdx.x;
  size_t base = (size_t)img * K;
  if (tid < 16) {
    int i = i0 + tid;
    const float* p4 = candObb + (base + i) * 4;
    s_ib[tid][0] = p4[0]; s_ib[tid][1] = p4[1];
    s_ib[tid][2] = p4[2]; s_ib[tid][3] = p4[3];
    s_ia[tid] = candArea[base + i];
  }
  if (tid < 128) {
    int j = j0 + tid;
    const float* p4 = candObb + (base + j) * 4;
    s_jb[tid][0] = p4[0]; s_jb[tid][1] = p4[1];
    s_jb[tid][2] = p4[2]; s_jb[tid][3] = p4[3];
    s_ja[tid] = candArea[base + j];
  }
  __syncthreads();

  int wv = tid >> 5, lane = tid & 31;
  int n = lane & 15, hi = lane >> 4;
  bool lo = lane < 16;  // lanes 16-31 hold K=8.. in A/B layouts -> zero there
  v16h a = {}; v16h b = {};
  a[0] = lo ? (_Float16)s_ia[n] : (_Float16)0.0f;            // A[m][0] = area_i(m)
  a[1] = lo ? (_Float16)1.0f   : (_Float16)0.0f;             // A[m][1] = 1
  b[0] = lo ? (_Float16)1.0f   : (_Float16)0.0f;             // B[0][n] = 1
  b[1] = lo ? (_Float16)s_ja[wv * 16 + n] : (_Float16)0.0f;  // B[1][n] = area_j(n)
  v8f cz = {};
  v8f S = __builtin_amdgcn_wmma_f32_16x16x32_f16(
      false, a, false, b, (short)0, cz, false, false);

  int jt = j0 + wv * 16;
  float jx1 = s_jb[wv * 16 + n][0], jy1 = s_jb[wv * 16 + n][1];
  float jx2 = s_jb[wv * 16 + n][2], jy2 = s_jb[wv * 16 + n][3];
#pragma unroll
  for (int r = 0; r < 8; ++r) {
    int ir = r + 8 * hi;  // C/D layout: lanes 0-15 -> M=r, lanes 16-31 -> M=r+8
    float ix1 = s_ib[ir][0], iy1 = s_ib[ir][1];
    float ix2 = s_ib[ir][2], iy2 = s_ib[ir][3];
    float iw = fmaxf(fminf(ix2, jx2) - fmaxf(ix1, jx1), 0.0f);
    float ih = fmaxf(fminf(iy2, jy2) - fmaxf(iy1, jy1), 0.0f);
    float inter = iw * ih;
    bool bit = inter * (1.0f + IOU_TH) > IOU_TH * S[r];
    unsigned long long bal = __ballot(bit);  // [15:0] -> row i0+r; [31:16] -> row i0+r+8
    if (lane == 0) {
      unsigned loBits = (unsigned)(bal & 0xFFFFull);
      unsigned hiBits = (unsigned)((bal >> 16) & 0xFFFFull);
      int shift = jt & 16;
      int wword = jt >> 5;
      if (loBits) atomicOr(&mask[(base + i0 + r) * 32 + wword], loBits << shift);
      if (hiBits) atomicOr(&mask[(base + i0 + r + 8) * 32 + wword], hiBits << shift);
    }
  }
}

// ----------------- stage 5: greedy resolve, one wave32 per image; lane owns 32 cols
__global__ __launch_bounds__(32)
void k_resolve(const unsigned* __restrict__ mask, const int* __restrict__ candValid,
               int* __restrict__ posOf) {
  int img = blockIdx.x;
  int lane = threadIdx.x;
  size_t base = (size_t)img * K;
  unsigned vm = 0;
  for (int b2 = 0; b2 < 32; ++b2)
    vm |= (candValid[base + lane * 32 + b2] ? 1u : 0u) << b2;
  unsigned removed = 0, keepM = 0;
  for (int i = 0; i < K; ++i) {
    int owner = i >> 5, bp = i & 31;
    int d = 0;
    if (lane == owner) d = (int)((vm >> bp) & 1u) & (int)((~(removed >> bp)) & 1u);
    d = __shfl(d, owner);
    if (d) {  // uniform across the wave after shfl
      if (lane == owner) keepM |= (1u << bp);
      unsigned row = mask[(base + i) * 32 + lane];
      unsigned m;
      if (lane < owner)      m = 0u;
      else if (lane > owner) m = 0xFFFFFFFFu;
      else                   m = (bp == 31) ? 0u : (0xFFFFFFFEu << bp);  // j > i only
      removed |= row & m;
    }
  }
  int pc = __popc(keepM);
  int before = 0;
  for (int l2 = 0; l2 < 32; ++l2) {
    int v = __shfl(pc, l2);
    if (l2 < lane) before += v;
  }
  for (int b2 = 0; b2 < 32; ++b2) {
    int j = lane * 32 + b2;
    int p = -1;
    if ((keepM >> b2) & 1u) {
      unsigned lowmask = (b2 == 0) ? 0u : (0xFFFFFFFFu >> (32 - b2));
      int rank = before + __popc(keepM & lowmask);
      if (rank < MAX_DET) p = rank;
    }
    posOf[base + j] = p;
  }
}

// ------------------ stage 6: scatter kept rows; logits gathered only for survivors
__global__ __launch_bounds__(256)
void k_write(const float* __restrict__ candBox, const float* __restrict__ candConf,
             const float* __restrict__ candCls, const int* __restrict__ candIdx,
             const int* __restrict__ posOf, const float* __restrict__ logits,
             float* __restrict__ out) {
  int grp = blockIdx.x * 8 + (threadIdx.x >> 5);
  int lane = threadIdx.x & 31;
  if (grp >= B * K) return;
  int img = grp >> 10;
  int pos = posOf[grp];
  if (pos < 0 || pos >= MAX_DET) return;
  int idx = candIdx[grp];
  if (idx < 0) return;
  size_t ob = ((size_t)img * MAX_DET + pos) * OUT_D;
  const float* lg = logits + ((size_t)img * N + idx) * NC;
  for (int e = lane; e < OUT_D; e += 32) {
    float v;
    if (e < 4)       v = candBox[grp * 4 + e];
    else if (e == 4) v = candConf[grp];
    else if (e == 5) v = candCls[grp];
    else             v = lg[e - 6];
    out[ob + e] = v;
  }
}

// ---------------------------------------------------------------------- launcher
extern "C" void kernel_launch(void* const* d_in, const int* in_sizes, int n_in,
                              void* d_out, int out_size, void* d_ws, size_t ws_size,
                              hipStream_t stream) {
  (void)in_sizes; (void)n_in; (void)out_size; (void)ws_size;
  const float* pred   = (const float*)d_in[0];
  const float* logits = (const float*)d_in[1];
  float* out = (float*)d_out;

  char* ws = (char*)d_ws;
  size_t off = 0;
  auto carve = [&](size_t bytes) -> void* {
    void* p = ws + off;
    off = (off + bytes + 255) & ~(size_t)255;
    return p;
  };
  float*    score     = (float*)carve((size_t)B * N * 4);          // 1.61 MB
  int*      candIdx   = (int*)carve((size_t)B * K * 4);
  float*    candBox   = (float*)carve((size_t)B * K * 4 * 4);
  float*    candObb   = (float*)carve((size_t)B * K * 4 * 4);
  float*    candArea  = (float*)carve((size_t)B * K * 4);
  float*    candConf  = (float*)carve((size_t)B * K * 4);
  float*    candCls   = (float*)carve((size_t)B * K * 4);
  int*      candValid = (int*)carve((size_t)B * K * 4);
  unsigned* mask      = (unsigned*)carve((size_t)B * K * 32 * 4);  // 2 MB bitmask
  int*      posOf     = (int*)carve((size_t)B * K * 4);
  // total ~4.7 MB of workspace

  k_zero<<<1024, 256, 0, stream>>>(out, mask, posOf);
  k_score<<<B * TILES_PER_IMG, ROWS_PER_TILE, 0, stream>>>(pred, score);
  k_topk<<<B, 256, 0, stream>>>(score, candIdx);
  k_gather<<<(B * K) / 256, 256, 0, stream>>>(pred, candIdx, candBox, candObb,
                                              candArea, candConf, candCls, candValid);
  k_mask<<<B * 512, 256, 0, stream>>>(candObb, candArea, mask);
  k_resolve<<<B, 32, 0, stream>>>(mask, candValid, posOf);
  k_write<<<(B * K) / 8, 256, 0, stream>>>(candBox, candConf, candCls, candIdx,
                                           posOf, logits, out);
}
